// TSAEOurs_33380485824838
// MI455X (gfx1250) — matrix operand order repaired and checked by reading, not compiled, not run on hardware
//
#include <hip/hip_runtime.h>
#include <hip/hip_bf16.h>
#include <math.h>

// ---------------------------------------------------------------------------
// Problem sizes (fixed by the reference)
// ---------------------------------------------------------------------------
#define D_INx   2304
#define D_SAEx  18432
#define KSEL    100
#define HH      9216
#define BBx     2048
#define NROWS   4096        // 2 * B   (prev/cur interleaved: row = b*2 + s)

typedef __attribute__((ext_vector_type(16))) __bf16 v16bf;
typedef __attribute__((ext_vector_type(8)))  float  v8f;

// ---------------------------------------------------------------------------
// bf16 WMMA GEMM:  C[M][N] = A[M][Kd] (bf16, row-major)
//                          @ Bt[N][Kd]^T (bf16, stored N-major = B transposed)
//                          + bias[N] (optional)
//
// Block tile 128x256, K-tile 32, double-buffered LDS filled with CDNA5
// GLOBAL_LOAD_ASYNC_TO_LDS_B128 (ASYNCcnt-tracked, no VGPR staging).
// 8 waves (wave32) in a 2x4 grid; each wave owns a 64x64 sub-tile =
// 4x4 WMMA 16x16 accumulators (v8f each).
// ---------------------------------------------------------------------------
#define MTm 128
#define NTn 256
#define KT  32

union FragU { v16bf v; uint4 q[2]; };

__device__ __forceinline__ void async_tile16(const __bf16* g, const __bf16* lds)
{
    // VDST = VGPR holding LDS byte address (low 32 bits of the generic
    // shared-aperture address per ISA 10.2), VADDR = 64-bit global address.
    unsigned           l  = (unsigned)(uintptr_t)lds;
    unsigned long long ga = (unsigned long long)(uintptr_t)g;
    asm volatile("global_load_async_to_lds_b128 %0, %1, off"
                 :: "v"(l), "v"(ga) : "memory");
}

__global__ __launch_bounds__(256)
void wmma_gemm_bf16(const __bf16* __restrict__ A,
                    const __bf16* __restrict__ Bt,
                    const float*  __restrict__ bias,
                    float*        __restrict__ C,
                    int M, int N, int Kd)
{
    __shared__ __align__(16) __bf16 As[2][MTm][KT];   // 2 x 8 KB
    __shared__ __align__(16) __bf16 Bs[2][NTn][KT];   // 2 x 16 KB

    const int tid   = threadIdx.x;
    const int lane  = tid & 31;
    const int wave  = tid >> 5;
    const int l16   = lane & 15;
    const int lhalf = lane >> 4;               // 0 / 1
    const int m0    = blockIdx.y * MTm;
    const int n0    = blockIdx.x * NTn;
    const int wm    = (wave >> 2) * 64;        // wave M offset (2 wave-rows)
    const int wn    = (wave & 3)  * 64;        // wave N offset (4 wave-cols)

    // accumulators: [mt 0..3][nt 0..3]
    v8f acc[4][4];
#pragma unroll
    for (int nt = 0; nt < 4; ++nt) {
        float bv = bias ? bias[n0 + wn + nt * 16 + l16] : 0.0f;
#pragma unroll
        for (int mt = 0; mt < 4; ++mt)
#pragma unroll
            for (int i = 0; i < 8; ++i) acc[mt][nt][i] = bv;
    }

    // async tile-loader mapping: 16-element (32 B) K-chunks
    const int arow = tid >> 1;                 // 0..127  (A rows)
    const int akc  = (tid & 1) * 16;           // 0 / 16
    const int brow0 = tid >> 1;                // B rows 0..127
    const int brow1 = brow0 + 128;             // B rows 128..255

    const int nst = Kd / KT;

    auto issue = [&](int stage, int buf) {
        const int k0 = stage * KT;
        async_tile16(A  + (size_t)(m0 + arow)  * Kd + k0 + akc, &As[buf][arow][akc]);
        async_tile16(Bt + (size_t)(n0 + brow0) * Kd + k0 + akc, &Bs[buf][brow0][akc]);
        async_tile16(Bt + (size_t)(n0 + brow1) * Kd + k0 + akc, &Bs[buf][brow1][akc]);
    };

    issue(0, 0);

    for (int s = 0; s < nst; ++s) {
        const int buf = s & 1;
        if (s + 1 < nst) {
            issue(s + 1, buf ^ 1);
            // drain the older group (3 async ops of stage s), keep 3 in flight
            asm volatile("s_wait_asynccnt 0x3" ::: "memory");
        } else {
            asm volatile("s_wait_asynccnt 0x0" ::: "memory");
        }
        __syncthreads();

        // A fragments: lane half h holds K[h*8 .. h*8+7] and K[h*8+16 .. +23]
        FragU af[4];
#pragma unroll
        for (int mt = 0; mt < 4; ++mt) {
            const __bf16* p = &As[buf][wm + mt * 16 + l16][lhalf * 8];
            af[mt].q[0] = *(const uint4*)p;
            af[mt].q[1] = *(const uint4*)(p + 16);
        }
        // B fragments: lane half h holds K[h*8 .. h*8+7] and K[h*8+16 .. +23]
        FragU bfr[4];
#pragma unroll
        for (int nt = 0; nt < 4; ++nt) {
            const __bf16* p = &Bs[buf][wn + nt * 16 + l16][lhalf * 8];
            bfr[nt].q[0] = *(const uint4*)p;
            bfr[nt].q[1] = *(const uint4*)(p + 16);
        }
#pragma unroll
        for (int mt = 0; mt < 4; ++mt)
#pragma unroll
            for (int nt = 0; nt < 4; ++nt)
                acc[mt][nt] = __builtin_amdgcn_wmma_f32_16x16x32_bf16(
                    false, af[mt].v, false, bfr[nt].v,
                    (short)0, acc[mt][nt], false, false);

        __syncthreads();   // protect buf before stage s+2 overwrites it
    }

    // Store: VGPR i -> row (mt*16 + i + lhalf*8), col (nt*16 + l16)
#pragma unroll
    for (int mt = 0; mt < 4; ++mt)
#pragma unroll
        for (int nt = 0; nt < 4; ++nt) {
            int col = n0 + wn + nt * 16 + l16;
#pragma unroll
            for (int i = 0; i < 8; ++i) {
                int row = m0 + wm + mt * 16 + i + lhalf * 8;
                C[(size_t)row * N + col] = acc[mt][nt][i];
            }
        }
}

// ---------------------------------------------------------------------------
// Conversion kernels (f32 -> bf16); W_enc is also transposed to [N][K] so the
// GEMM's B operand is K-contiguous.
// ---------------------------------------------------------------------------
__global__ __launch_bounds__(256)
void convert_wenc(const float* __restrict__ W, __bf16* __restrict__ WT)
{
    size_t i = (size_t)blockIdx.x * 256 + threadIdx.x;     // over D_IN * D_SAE
    if (i >= (size_t)D_INx * D_SAEx) return;
    int k = (int)(i / D_SAEx);
    int n = (int)(i % D_SAEx);
    WT[(size_t)n * D_INx + k] = (__bf16)W[i];
}

__global__ __launch_bounds__(256)
void convert_x(const float* __restrict__ x, __bf16* __restrict__ xb)
{
    size_t i = (size_t)blockIdx.x * 256 + threadIdx.x;     // over NROWS * D_IN
    if (i >= (size_t)NROWS * D_INx) return;
    xb[i] = (__bf16)x[i];
}

// ---------------------------------------------------------------------------
// Top-K(100) per row via iterative argmax in LDS (ties -> lowest index, like
// jax.lax.top_k).  Row r: s = r&1 (0=prev, 1=cur), b = r>>1.
// cur rows additionally write the dense (relu'd) z row into d_out.
// ---------------------------------------------------------------------------
__global__ __launch_bounds__(256)
void topk_kernel(const float* __restrict__ pre,
                 float* __restrict__ z_out,
                 int*   __restrict__ cur_idx, float* __restrict__ cur_val,
                 int*   __restrict__ prev_idx, float* __restrict__ prev_val)
{
    __shared__ float vals[D_SAEx];     // 72 KB (<= 320 KB/WGP on CDNA5)
    __shared__ float rmax[256];
    __shared__ int   rarg[256];
    __shared__ int   kidx[KSEL];
    __shared__ float kval[KSEL];

    const int row = blockIdx.x;
    const int tid = threadIdx.x;
    const int s   = row & 1;
    const int b   = row >> 1;

    for (int j = tid; j < D_SAEx; j += 256)
        vals[j] = pre[(size_t)row * D_SAEx + j];
    __syncthreads();

    int*   oidx = s ? (cur_idx + b * KSEL) : (prev_idx + b * KSEL);
    float* oval = s ? (cur_val + b * KSEL) : (prev_val + b * KSEL);

    for (int k = 0; k < KSEL; ++k) {
        float best = -INFINITY; int bi = 0x7fffffff;
        for (int j = tid; j < D_SAEx; j += 256) {
            float v = vals[j];
            if (v > best || (v == best && j < bi)) { best = v; bi = j; }
        }
        rmax[tid] = best; rarg[tid] = bi;
        __syncthreads();
        for (int off = 128; off > 0; off >>= 1) {
            if (tid < off) {
                float vo = rmax[tid + off]; int io = rarg[tid + off];
                if (vo > rmax[tid] || (vo == rmax[tid] && io < rarg[tid])) {
                    rmax[tid] = vo; rarg[tid] = io;
                }
            }
            __syncthreads();
        }
        if (tid == 0) {
            int   wi = rarg[0];
            float wv = rmax[0] > 0.0f ? rmax[0] : 0.0f;    // relu(top_k values)
            kidx[k] = wi;  kval[k] = wv;
            oidx[k] = wi;  oval[k] = wv;
            vals[wi] = -INFINITY;
        }
        __syncthreads();
    }

    if (s) {  // cur: dense z row into output
        float* zr = z_out + (size_t)b * D_SAEx;
        for (int j = tid; j < D_SAEx; j += 256) zr[j] = 0.0f;
        __syncthreads();
        if (tid < KSEL) zr[kidx[tid]] = kval[tid];
    }
}

// ---------------------------------------------------------------------------
// L2-normalize z[:, :H] (sparse) and scatter into dense bf16 rows for the
// contrastive GEMM.  One block per (row, side).
// ---------------------------------------------------------------------------
__global__ __launch_bounds__(256)
void norm_scatter(const int* __restrict__ cur_idx, const float* __restrict__ cur_val,
                  const int* __restrict__ prev_idx, const float* __restrict__ prev_val,
                  __bf16* __restrict__ zaH, __bf16* __restrict__ zbH)
{
    const int r = blockIdx.x, tid = threadIdx.x;
    const int s = r & 1, b = r >> 1;
    const int*   idx = (s ? cur_idx : prev_idx) + b * KSEL;
    const float* val = (s ? cur_val : prev_val) + b * KSEL;
    __bf16* dense = (s ? zaH : zbH) + (size_t)b * HH;

    __shared__ float red[256];
    __shared__ float snorm;
    float loc = 0.0f;
    if (tid < KSEL) { int i = idx[tid]; float v = val[tid]; if (i < HH) loc = v * v; }
    red[tid] = loc;
    __syncthreads();
    for (int off = 128; off > 0; off >>= 1) {
        if (tid < off) red[tid] += red[tid + off];
        __syncthreads();
    }
    if (tid == 0) snorm = fmaxf(sqrtf(red[0]), 1e-8f);
    __syncthreads();

    for (int j = tid; j < HH; j += 256) dense[j] = (__bf16)0.0f;
    __syncthreads();
    if (tid < KSEL) { int i = idx[tid]; if (i < HH) dense[i] = (__bf16)(val[tid] / snorm); }
}

// ---------------------------------------------------------------------------
// Sparse decoder: x_hat = z_cur @ W_dec + b_dec (100 nnz per row), plus the
// restricted x_hat_h (idx < H) and the two squared-error sums.
// One block per batch row; thread t owns columns {t, t+256, ..., t+8*256}.
// ---------------------------------------------------------------------------
__global__ __launch_bounds__(256)
void decoder_kernel(const int* __restrict__ cur_idx, const float* __restrict__ cur_val,
                    const float* __restrict__ W_dec, const float* __restrict__ b_dec,
                    const float* __restrict__ x,
                    float* __restrict__ xhat_out, float* __restrict__ accum)
{
    const int b = blockIdx.x, tid = threadIdx.x;
    __shared__ int   sidx[KSEL];
    __shared__ float sval[KSEL];
    __shared__ float redf[256], redh[256];
    if (tid < KSEL) { sidx[tid] = cur_idx[b * KSEL + tid]; sval[tid] = cur_val[b * KSEL + tid]; }
    __syncthreads();

    float af[9], ah[9];
#pragma unroll
    for (int c = 0; c < 9; ++c) {
        float bd = b_dec[tid + c * 256];
        af[c] = bd; ah[c] = bd;
    }
    for (int j = 0; j < KSEL; ++j) {
        int   fi = sidx[j];
        float v  = sval[j];
        const float* wr = W_dec + (size_t)fi * D_INx;
        bool inH = fi < HH;
#pragma unroll
        for (int c = 0; c < 9; ++c) {
            float wv = wr[tid + c * 256];
            af[c] += v * wv;
            if (inH) ah[c] += v * wv;
        }
    }
    const float* xc = x + ((size_t)b * 2 + 1) * D_INx;   // x_cur row
    float sf = 0.0f, sh = 0.0f;
#pragma unroll
    for (int c = 0; c < 9; ++c) {
        int col = tid + c * 256;
        float xv = xc[col];
        float df = af[c] - xv;  sf += df * df;
        float dh = ah[c] - xv;  sh += dh * dh;
        xhat_out[(size_t)b * D_INx + col] = af[c];
    }
    redf[tid] = sf; redh[tid] = sh;
    __syncthreads();
    for (int off = 128; off > 0; off >>= 1) {
        if (tid < off) { redf[tid] += redf[tid + off]; redh[tid] += redh[tid + off]; }
        __syncthreads();
    }
    if (tid == 0) {
        atomicAdd(&accum[0], redh[0]);   // l_high numerator
        atomicAdd(&accum[1], redf[0]);   // l_full numerator
    }
}

// ---------------------------------------------------------------------------
// Cross-entropy over sim rows / columns: accumulate -log_softmax(sim)[i,i].
// ---------------------------------------------------------------------------
__global__ __launch_bounds__(256)
void ce_kernel(const float* __restrict__ sim, float* __restrict__ accum,
               int transposed, int slot)
{
    const int i = blockIdx.x, tid = threadIdx.x;
    __shared__ float red[256];
    __shared__ float smax, ssum;

    float m = -INFINITY;
    for (int j = tid; j < BBx; j += 256) {
        float v = transposed ? sim[(size_t)j * BBx + i] : sim[(size_t)i * BBx + j];
        m = fmaxf(m, v);
    }
    red[tid] = m; __syncthreads();
    for (int off = 128; off > 0; off >>= 1) {
        if (tid < off) red[tid] = fmaxf(red[tid], red[tid + off]);
        __syncthreads();
    }
    if (tid == 0) smax = red[0];
    __syncthreads();

    float sum = 0.0f;
    for (int j = tid; j < BBx; j += 256) {
        float v = transposed ? sim[(size_t)j * BBx + i] : sim[(size_t)i * BBx + j];
        sum += __expf(v - smax);
    }
    red[tid] = sum; __syncthreads();
    for (int off = 128; off > 0; off >>= 1) {
        if (tid < off) red[tid] += red[tid + off];
        __syncthreads();
    }
    if (tid == 0) {
        ssum = red[0];
        float lsm = sim[(size_t)i * BBx + i] - smax - __logf(ssum);
        atomicAdd(&accum[slot], -lsm);
    }
}

__global__ void zero_accum(float* accum)
{
    if (threadIdx.x < 4) accum[threadIdx.x] = 0.0f;
}

__global__ void finalize_kernel(const float* __restrict__ accum, float* __restrict__ out)
{
    if (threadIdx.x == 0) {
        float l_matr  = (accum[0] + accum[1]) / ((float)BBx * (float)D_INx);
        float l_contr = 0.5f * (accum[2] + accum[3]) / (float)BBx;
        out[0] = l_matr + l_contr;
    }
}

// ---------------------------------------------------------------------------
// Launcher
// ---------------------------------------------------------------------------
extern "C" void kernel_launch(void* const* d_in, const int* in_sizes, int n_in,
                              void* d_out, int out_size, void* d_ws, size_t ws_size,
                              hipStream_t stream)
{
    const float* x     = (const float*)d_in[0];
    const float* W_enc = (const float*)d_in[1];
    const float* W_dec = (const float*)d_in[2];
    const float* b_enc = (const float*)d_in[3];
    const float* b_dec = (const float*)d_in[4];
    float* out = (float*)d_out;

    // workspace layout (bytes)
    char* w = (char*)d_ws;
    const size_t OFF_WT  = 0;                                      // bf16 [D_SAE][D_IN]
    const size_t OFF_XB  = OFF_WT  + (size_t)D_SAEx * D_INx * 2;   // bf16 [4096][D_IN]
    const size_t OFF_PRE = OFF_XB  + (size_t)NROWS * D_INx * 2;    // f32  [4096][D_SAE]
    // zaH/zbH/sim alias the pre buffer (pre fully consumed by topk first)
    const size_t OFF_ZA  = OFF_PRE;                                // bf16 [B][H]
    const size_t OFF_ZB  = OFF_ZA  + (size_t)BBx * HH * 2;         // bf16 [B][H]
    const size_t OFF_SIM = OFF_ZB  + (size_t)BBx * HH * 2;         // f32  [B][B]
    const size_t OFF_CI  = OFF_PRE + (size_t)NROWS * D_SAEx * 4;   // int  [B][K]
    const size_t OFF_CV  = OFF_CI  + (size_t)BBx * KSEL * 4;       // f32
    const size_t OFF_PI  = OFF_CV  + (size_t)BBx * KSEL * 4;       // int
    const size_t OFF_PV  = OFF_PI  + (size_t)BBx * KSEL * 4;       // f32
    const size_t OFF_ACC = OFF_PV  + (size_t)BBx * KSEL * 4;       // f32 [4]

    __bf16* WT   = (__bf16*)(w + OFF_WT);
    __bf16* xb   = (__bf16*)(w + OFF_XB);
    float*  pre  = (float*) (w + OFF_PRE);
    __bf16* zaH  = (__bf16*)(w + OFF_ZA);
    __bf16* zbH  = (__bf16*)(w + OFF_ZB);
    float*  sim  = (float*) (w + OFF_SIM);
    int*    ci   = (int*)   (w + OFF_CI);
    float*  cv   = (float*) (w + OFF_CV);
    int*    pi   = (int*)   (w + OFF_PI);
    float*  pv   = (float*) (w + OFF_PV);
    float*  acc  = (float*) (w + OFF_ACC);

    float* xhat_out = out + 1;
    float* z_out    = out + 1 + (size_t)BBx * D_INx;

    zero_accum<<<1, 32, 0, stream>>>(acc);

    // f32 -> bf16 (W_enc transposed so GEMM B operand is K-contiguous)
    {
        size_t n = (size_t)D_INx * D_SAEx;
        convert_wenc<<<(unsigned)((n + 255) / 256), 256, 0, stream>>>(W_enc, WT);
    }
    {
        size_t n = (size_t)NROWS * D_INx;
        convert_x<<<(unsigned)((n + 255) / 256), 256, 0, stream>>>(x, xb);
    }

    // encoder: pre[4096][18432] = xb @ W_enc + b_enc
    wmma_gemm_bf16<<<dim3(D_SAEx / NTn, NROWS / MTm), 256, 0, stream>>>(
        xb, WT, b_enc, pre, NROWS, D_SAEx, D_INx);

    // top-k + relu; dense z_cur into d_out
    topk_kernel<<<NROWS, 256, 0, stream>>>(pre, z_out, ci, cv, pi, pv);

    // l2-normalize z[:, :H], scatter into dense bf16 rows
    norm_scatter<<<NROWS, 256, 0, stream>>>(ci, cv, pi, pv, zaH, zbH);

    // sim[2048][2048] = zaH @ zbH^T   (zbH stored row-major == N-major B)
    wmma_gemm_bf16<<<dim3(BBx / NTn, BBx / MTm), 256, 0, stream>>>(
        zaH, zbH, nullptr, sim, BBx, BBx, HH);

    // sparse decoder + reconstruction losses; x_hat into d_out
    decoder_kernel<<<BBx, 256, 0, stream>>>(ci, cv, W_dec, b_dec, x, xhat_out, acc);

    // contrastive CE (rows and columns)
    ce_kernel<<<BBx, 256, 0, stream>>>(sim, acc, 0, 2);
    ce_kernel<<<BBx, 256, 0, stream>>>(sim, acc, 1, 3);

    finalize_kernel<<<1, 32, 0, stream>>>(acc, out);
}
